// UResNet_67095979098162
// MI455X (gfx1250) — matrix operand order repaired and checked by reading, not compiled, not run on hardware
//
#include <hip/hip_runtime.h>

// ---------------------------------------------------------------------------
// CDNA5 (gfx1250) UResNet forward.
// Submanifold 3^3 convs as gather -> v_wmma_f32_16x16x32_bf16 GEMMs with
// fragment-native repacked weights and padded bf16 feature staging
// (no LDS, no predication in the hot loop).
// ---------------------------------------------------------------------------

typedef __attribute__((ext_vector_type(16))) __bf16 v16bf;
typedef __attribute__((ext_vector_type(8)))  __bf16 v8bf;
typedef __attribute__((ext_vector_type(8)))  float  v8f;

// ---------------------------------------------------------------------------
// Submanifold conv: out[n, :] = sum_k Xb[nbr[n,k]] @ W[k]  (+ optional add)
// Xb: [N+1][ldx] bf16, BN+ReLU pre-applied, K-padded, row N == zeros.
// Wr: repacked [27][nKT][NCT][32][16] bf16 fragments (zero-padded K).
// One wave per block = one 16-row M tile; NCT column tiles in registers.
// ---------------------------------------------------------------------------
template <int NCT>
__global__ __launch_bounds__(32) void subm_wmma_kernel(
    const __bf16* __restrict__ Xb, int ldx,
    const int*    __restrict__ nbr,              // [N][27]
    const __bf16* __restrict__ Wr,
    const float*  __restrict__ add, int ld_add,  // residual or null
    float* __restrict__ out, int ld_out,
    int N, int nKT)
{
    const int lane = threadIdx.x & 31;
    const int half = lane >> 4;
    const int r    = lane & 15;
    const int row0 = blockIdx.x * 16;
    const int myrow = row0 + r;
    const bool rowok = myrow < N;
    const int* nrow = nbr + (long)(rowok ? myrow : 0) * 27;

    v8f acc[NCT];
#pragma unroll
    for (int ct = 0; ct < NCT; ++ct)
#pragma unroll
        for (int e = 0; e < 8; ++e) acc[ct][e] = 0.0f;

    for (int k = 0; k < 27; ++k) {
        int j = rowok ? nrow[k] : -1;
        if (j < 0) j = N;                        // zero row => masked neighbor
        const __bf16* xr = Xb + (long)j * ldx;
        const __bf16* wk = Wr + ((long)k * nKT * NCT) * 512 + lane * 16;

        for (int kt = 0; kt < nKT; ++kt) {
            // A fragment (16x32 bf16): lane holds row r; two aligned 16B loads.
            // half=0: K = kt*32 + {0..7} and {16..23}; half=1: +8.
            const v8bf lo = *(const v8bf*)(xr + kt * 32 + 8 * half);
            const v8bf hi = *(const v8bf*)(xr + kt * 32 + 16 + 8 * half);
            v16bf a;
#pragma unroll
            for (int e = 0; e < 8; ++e) { a[e] = lo[e]; a[8 + e] = hi[e]; }

#pragma unroll
            for (int ct = 0; ct < NCT; ++ct) {
                // B fragment: one contiguous 32B load per lane (repacked).
                const v16bf b = *(const v16bf*)(wk + ((long)kt * NCT + ct) * 512);
                acc[ct] = __builtin_amdgcn_wmma_f32_16x16x32_bf16(
                    false, a, false, b, (short)0, acc[ct], false, false);
            }
        }
    }

    // C/D layout: VGPR v -> row = v + 8*half, col = ct*16 + (lane&15)
#pragma unroll
    for (int ct = 0; ct < NCT; ++ct) {
        const int col = ct * 16 + r;
#pragma unroll
        for (int v = 0; v < 8; ++v) {
            const int row = row0 + v + 8 * half;
            if (row < N) {
                float o = acc[ct][v];
                if (add) o += add[(long)row * ld_add + col];
                out[(long)row * ld_out + col] = o;
            }
        }
    }
}

// ---------------------------------------------------------------------------
// Stage a subm input: y[N+1][ldy] bf16 = bnrelu(x) (zero pad cols, zero row N)
// ---------------------------------------------------------------------------
__global__ void stage_bf16_kernel(const float* __restrict__ x, int ld_in,
                                  const float* __restrict__ s, const float* __restrict__ b,
                                  __bf16* __restrict__ y, int ldy, int N, int Cin) {
    long t = (long)blockIdx.x * blockDim.x + threadIdx.x;
    long tot = (long)(N + 1) * ldy;
    if (t >= tot) return;
    long i = t / ldy; int c = (int)(t - i * ldy);
    float v = 0.0f;
    if (i < N && c < Cin) {
        v = x[i * (long)ld_in + c];
        if (s) { v = v * s[c] + b[c]; v = v > 0.0f ? v : 0.0f; }
    }
    y[t] = (__bf16)v;
}

// ---------------------------------------------------------------------------
// Repack conv weights [27][Cin][Cout] f32 -> fragment-native bf16
// out[k][kt][ct][lane][e], K zero-padded to nKT*32.
// ---------------------------------------------------------------------------
__global__ void repack_w_kernel(const float* __restrict__ W,
                                __bf16* __restrict__ out,
                                int Cin, int Cout) {
    int nKT = ((Cin + 31) & ~31) >> 5;
    int nCT = Cout >> 4;
    long tot = (long)27 * nKT * nCT * 512;
    long t = (long)blockIdx.x * blockDim.x + threadIdx.x;
    if (t >= tot) return;
    int e    = (int)(t & 15);
    int lane = (int)((t >> 4) & 31);
    long rest = t >> 9;
    int ct = (int)(rest % nCT); rest /= nCT;
    int kt = (int)(rest % nKT);
    int k  = (int)(rest / nKT);
    int half = lane >> 4, r = lane & 15;
    int jv = e >> 1, p = e & 1;
    int K = kt * 32 + 8 * half + (e & 7) + 16 * (e >> 3);
    int col = ct * 16 + r;
    float v = (K < Cin) ? W[((long)k * Cin + K) * Cout + col] : 0.0f;
    out[t] = (__bf16)v;
}

// ---------------------------------------------------------------------------
// Small helper kernels (bandwidth-trivial parts of the network).
// ---------------------------------------------------------------------------
__global__ void zero_kernel(float* __restrict__ p, long n) {
    long i = (long)blockIdx.x * blockDim.x + threadIdx.x;
    if (i < n) p[i] = 0.0f;
}

// C[:, :a] = x  (row stride 2a on destination)
__global__ void copy_strided_kernel(const float* __restrict__ x, float* __restrict__ C,
                                    long N, int a) {
    long t = (long)blockIdx.x * blockDim.x + threadIdx.x;
    if (t >= N * a) return;
    long i = t / a; int c = (int)(t - i * a);
    C[i * 2 * (long)a + c] = x[t];
}

// stride-2 conv: y[parent[i]] += bnrelu(x[i]) @ W[offset[i]]   (segment_sum)
__global__ void down_kernel(const float* __restrict__ x,
                            const float* __restrict__ s, const float* __restrict__ b,
                            const int* __restrict__ parent, const int* __restrict__ offs,
                            const float* __restrict__ W,   // [8][Cin][Cout] f32
                            float* __restrict__ y, int Nf, int Cin, int Cout) {
    int i  = blockIdx.x * blockDim.x + threadIdx.x;
    int co = blockIdx.y;
    if (i >= Nf) return;
    const float* Wk = W + ((long)offs[i] * Cin) * Cout + co;
    const float* xi = x + (long)i * Cin;
    float acc = 0.0f;
    for (int ci = 0; ci < Cin; ++ci) {
        float v = xi[ci] * s[ci] + b[ci];
        v = v > 0.0f ? v : 0.0f;
        acc += v * Wk[(long)ci * Cout];
    }
    atomicAdd(&y[(long)parent[i] * Cout + co], acc);
}

// stride-2 deconv: y[i, colOff+:] = bnrelu(coarse[parent[i]]) @ W[offset[i]]
__global__ void up_kernel(const float* __restrict__ xc,
                          const float* __restrict__ s, const float* __restrict__ b,
                          const int* __restrict__ parent, const int* __restrict__ offs,
                          const float* __restrict__ W,    // [8][Cin][Cout] f32
                          float* __restrict__ y, int ld_y, int Nf, int Cin, int Cout) {
    int i  = blockIdx.x * blockDim.x + threadIdx.x;
    int co = blockIdx.y;
    if (i >= Nf) return;
    const float* Wk = W + ((long)offs[i] * Cin) * Cout + co;
    const float* xi = xc + (long)parent[i] * Cin;
    float acc = 0.0f;
    for (int ci = 0; ci < Cin; ++ci) {
        float v = xi[ci] * s[ci] + b[ci];
        v = v > 0.0f ? v : 0.0f;
        acc += v * Wk[(long)ci * Cout];
    }
    y[(long)i * ld_y + co] = acc;
}

// NetworkInNetwork shortcut: y = x @ W   (W: [Cin][Cout] f32)
__global__ void nin_kernel(const float* __restrict__ x, int ld_x,
                           const float* __restrict__ W,
                           float* __restrict__ y, int ld_y, int N, int Cin, int Cout) {
    int i  = blockIdx.x * blockDim.x + threadIdx.x;
    int co = blockIdx.y;
    if (i >= N) return;
    float acc = 0.0f;
    for (int ci = 0; ci < Cin; ++ci)
        acc += x[(long)i * ld_x + ci] * W[(long)ci * Cout + co];
    y[(long)i * ld_y + co] = acc;
}

// bn_out + relu -> lin1(16->8) + relu -> lin2(8->1)
__global__ void head_kernel(const float* __restrict__ x,
                            const float* __restrict__ s, const float* __restrict__ b,
                            const float* __restrict__ w1, const float* __restrict__ b1,
                            const float* __restrict__ w2, const float* __restrict__ b2,
                            float* __restrict__ out, int N) {
    int i = blockIdx.x * blockDim.x + threadIdx.x;
    if (i >= N) return;
    float h[16];
#pragma unroll
    for (int c = 0; c < 16; ++c) {
        float v = x[(long)i * 16 + c] * s[c] + b[c];
        h[c] = v > 0.0f ? v : 0.0f;
    }
    float o = b2[0];
#pragma unroll
    for (int j = 0; j < 8; ++j) {
        float g = b1[j];
#pragma unroll
        for (int c = 0; c < 16; ++c) g += h[c] * w1[c * 8 + j];
        g = g > 0.0f ? g : 0.0f;
        o += g * w2[j];
    }
    out[i] = o;
}

// ---------------------------------------------------------------------------
// Host orchestration
// ---------------------------------------------------------------------------
extern "C" void kernel_launch(void* const* d_in, const int* in_sizes, int n_in,
                              void* d_out, int out_size, void* d_ws, size_t ws_size,
                              hipStream_t stream) {
    (void)n_in; (void)out_size; (void)ws_size;
    const float* feats = (const float*)d_in[0];
    const float* prm   = (const float*)d_in[1];   // flattened params pytree (f32)
    const int* nbr[5];
    for (int l = 0; l < 5; ++l) nbr[l] = (const int*)d_in[2 + l];
    const int* parent[4]; const int* offs[4];
    for (int l = 0; l < 4; ++l) { parent[l] = (const int*)d_in[7 + l]; offs[l] = (const int*)d_in[11 + l]; }
    int Nl[5];
    for (int l = 0; l < 5; ++l) Nl[l] = in_sizes[2 + l] / 27;
    const int planes[5] = {16, 32, 48, 64, 80};

    // ---- parameter offsets: jax tree_flatten order (dict keys sorted) ----
    long off = 0;
    auto take = [&](long n) { long o = off; off += n; return o; };
    long roff = 0;   // repacked-weight blob element offset
    auto rtake = [&](long Cin, long Cout) {
        long nKT = ((Cin + 31) & ~31) / 32;
        long sz = 27 * nKT * (Cout / 16) * 512;
        long o = roff; roff += sz; return o;
    };
    struct Blk { long b1, b2, nin, s1, s2, w1, w2, rw1, rw2; };
    long bnout_b = take(16), bnout_s = take(16);
    Blk dec[4][2];
    for (int l = 0; l < 4; ++l) {
        long a = planes[l];
        Blk& B0 = dec[l][0];                 // block(2a -> a), has nin
        B0.b1 = take(2*a); B0.b2 = take(a); B0.nin = take(2*a*a);
        B0.s1 = take(2*a); B0.s2 = take(a);
        B0.w1 = take(27*2*a*a); B0.w2 = take(27*a*a);
        B0.rw1 = rtake(2*a, a); B0.rw2 = rtake(a, a);
        Blk& B1 = dec[l][1];                 // block(a -> a)
        B1.b1 = take(a); B1.b2 = take(a); B1.nin = -1;
        B1.s1 = take(a); B1.s2 = take(a);
        B1.w1 = take(27*a*a); B1.w2 = take(27*a*a);
        B1.rw1 = rtake(a, a); B1.rw2 = rtake(a, a);
    }
    long dwn_b[4], dwn_s[4], dwn_w[4];
    for (int l = 0; l < 4; ++l) {
        long a = planes[l], bp = planes[l+1];
        dwn_b[l] = take(a); dwn_s[l] = take(a); dwn_w[l] = take(8*a*bp);
    }
    Blk enc[5][2];
    for (int l = 0; l < 5; ++l) {
        long a = planes[l];
        for (int j = 0; j < 2; ++j) {
            Blk& B = enc[l][j];
            B.b1 = take(a); B.b2 = take(a); B.nin = -1;
            B.s1 = take(a); B.s2 = take(a);
            B.w1 = take(27*a*a); B.w2 = take(27*a*a);
            B.rw1 = rtake(a, a); B.rw2 = rtake(a, a);
        }
    }
    long lin1_b = take(8), lin1_w = take(16*8), lin2_b = take(1), lin2_w = take(8);
    long up_b[4], up_s[4], up_w[4];
    for (int l = 0; l < 4; ++l) {
        long a = planes[l], bp = planes[l+1];
        up_b[l] = take(bp); up_s[l] = take(bp); up_w[l] = take(8*bp*a);
    }
    long w_in = take(27*6*16);
    long rw_in = rtake(6, 16);
    const long RTOT = roff;

    // ---- workspace layout ----
    char* ws = (char*)d_ws;
    size_t wo = 0;
    auto walloc = [&](size_t bytes) {
        size_t r = wo; wo += (bytes + 255) & ~(size_t)255; return r;
    };
    __bf16* Wrep = (__bf16*)(ws + walloc((size_t)RTOT * 2));
    float* X[5];
    size_t maxN = 0;
    for (int l = 0; l < 5; ++l) {
        X[l] = (float*)(ws + walloc((size_t)Nl[l] * planes[l] * 4));
        if ((size_t)Nl[l] > maxN) maxN = Nl[l];
    }
    size_t maxC = 0, maxS = 0;
    for (int l = 0; l < 5; ++l) {
        size_t s1 = (size_t)Nl[l] * planes[l];
        if (s1 > maxS) maxS = s1;
        size_t c1 = (l < 4) ? (size_t)Nl[l] * 2 * planes[l] : s1;
        if (c1 > maxC) maxC = c1;
    }
    float*  Cb = (float*)(ws + walloc(maxC * 4));            // concat / conv scratch
    float*  Sb = (float*)(ws + walloc(maxS * 4));            // shortcut / ping-pong
    __bf16* Xs = (__bf16*)(ws + walloc((maxN + 1) * 128 * 2)); // staged bf16 input

    // ---- launch helpers ----
    auto repack = [&](long w_off, long r_off, int Cin, int Cout) {
        int nKT = ((Cin + 31) & ~31) >> 5;
        long tot = (long)27 * nKT * (Cout / 16) * 512;
        repack_w_kernel<<<(unsigned)((tot + 255) / 256), 256, 0, stream>>>(
            prm + w_off, Wrep + r_off, Cin, Cout);
    };
    auto subm = [&](const float* in, int ld_in, long so, long bo,
                    const int* nb, long r_off, const float* add, int ld_add,
                    float* outp, int ld_out, int N, int Cin, int Cout) {
        int cinPad = (Cin + 31) & ~31, nKT = cinPad >> 5;
        long tot = (long)(N + 1) * cinPad;
        stage_bf16_kernel<<<(unsigned)((tot + 255) / 256), 256, 0, stream>>>(
            in, ld_in, (so < 0) ? nullptr : prm + so, (bo < 0) ? nullptr : prm + bo,
            Xs, cinPad, N, Cin);
        dim3 g((unsigned)((N + 15) / 16));
        const __bf16* wp = Wrep + r_off;
        switch (Cout / 16) {
            case 1: subm_wmma_kernel<1><<<g,32,0,stream>>>(Xs,cinPad,nb,wp,add,ld_add,outp,ld_out,N,nKT); break;
            case 2: subm_wmma_kernel<2><<<g,32,0,stream>>>(Xs,cinPad,nb,wp,add,ld_add,outp,ld_out,N,nKT); break;
            case 3: subm_wmma_kernel<3><<<g,32,0,stream>>>(Xs,cinPad,nb,wp,add,ld_add,outp,ld_out,N,nKT); break;
            case 4: subm_wmma_kernel<4><<<g,32,0,stream>>>(Xs,cinPad,nb,wp,add,ld_add,outp,ld_out,N,nKT); break;
            default: subm_wmma_kernel<5><<<g,32,0,stream>>>(Xs,cinPad,nb,wp,add,ld_add,outp,ld_out,N,nKT); break;
        }
    };

    // 0) repack all submanifold conv weights into fragment-native bf16
    repack(w_in, rw_in, 6, 16);
    for (int l = 0; l < 5; ++l) {
        int a = planes[l];
        for (int j = 0; j < 2; ++j) {
            repack(enc[l][j].w1, enc[l][j].rw1, a, a);
            repack(enc[l][j].w2, enc[l][j].rw2, a, a);
        }
    }
    for (int l = 0; l < 4; ++l) {
        int a = planes[l];
        repack(dec[l][0].w1, dec[l][0].rw1, 2 * a, a);
        repack(dec[l][0].w2, dec[l][0].rw2, a, a);
        repack(dec[l][1].w1, dec[l][1].rw1, a, a);
        repack(dec[l][1].w2, dec[l][1].rw2, a, a);
    }

    // 1) input conv: x0 = subm(feats, nbr0, w_in)  (Cin=6 -> 16, no BN)
    subm(feats, 6, -1, -1, nbr[0], rw_in, nullptr, 0, X[0], 16, Nl[0], 6, 16);

    // 2) encoder sweep
    for (int l = 0; l < 5; ++l) {
        int a = planes[l], N = Nl[l];
        for (int j = 0; j < 2; ++j) {
            const Blk& B = enc[l][j];
            const float* in = (j == 0) ? X[l] : Sb;
            float*      op = (j == 0) ? Sb   : X[l];
            // t = subm(bnrelu(in), w1);  out = subm(bnrelu(t), w2) + in
            subm(in, a, B.s1, B.b1, nbr[l], B.rw1, nullptr, 0, Cb, a, N, a, a);
            subm(Cb, a, B.s2, B.b2, nbr[l], B.rw2, in, a, op, a, N, a, a);
        }
        if (l < 4) {
            int bp = planes[l+1], Nc = Nl[l+1];
            long nz = (long)Nc * bp;
            zero_kernel<<<(unsigned)((nz + 255) / 256), 256, 0, stream>>>(X[l+1], nz);
            dim3 g((unsigned)((N + 255) / 256), (unsigned)bp);
            down_kernel<<<g, 256, 0, stream>>>(X[l], prm + dwn_s[l], prm + dwn_b[l],
                                               parent[l], offs[l], prm + dwn_w[l],
                                               X[l+1], N, a, bp);
        }
    }

    // 3) decoder sweep
    for (int l = 3; l >= 0; --l) {
        int a = planes[l], bp = planes[l+1], N = Nl[l];
        {   // u = up(bnrelu(x_{l+1}))  -> C[:, a:]
            dim3 g((unsigned)((N + 255) / 256), (unsigned)a);
            up_kernel<<<g, 256, 0, stream>>>(X[l+1], prm + up_s[l], prm + up_b[l],
                                             parent[l], offs[l], prm + up_w[l],
                                             Cb + a, 2 * a, N, bp, a);
        }
        {   // C[:, :a] = x_l   (JoinTable)
            long nt = (long)N * a;
            copy_strided_kernel<<<(unsigned)((nt + 255) / 256), 256, 0, stream>>>(X[l], Cb, (long)N, a);
        }
        {   // dec block0: 2a -> a (nin shortcut)
            const Blk& B = dec[l][0];
            dim3 g((unsigned)((N + 255) / 256), (unsigned)a);
            nin_kernel<<<g, 256, 0, stream>>>(Cb, 2 * a, prm + B.nin, Sb, a, N, 2 * a, a);
            subm(Cb, 2 * a, B.s1, B.b1, nbr[l], B.rw1, nullptr, 0, X[l], a, N, 2 * a, a);
            subm(X[l], a, B.s2, B.b2, nbr[l], B.rw2, Sb, a, Cb, 2 * a, N, a, a);
        }
        {   // dec block1: a -> a (identity shortcut = C[:, :a])
            const Blk& B = dec[l][1];
            subm(Cb, 2 * a, B.s1, B.b1, nbr[l], B.rw1, nullptr, 0, Sb, a, N, a, a);
            subm(Sb, a, B.s2, B.b2, nbr[l], B.rw2, Cb, 2 * a, X[l], a, N, a, a);
        }
    }

    // 4) output head
    head_kernel<<<(unsigned)((Nl[0] + 255) / 256), 256, 0, stream>>>(
        X[0], prm + bnout_s, prm + bnout_b,
        prm + lin1_w, prm + lin1_b, prm + lin2_w, prm + lin2_b,
        (float*)d_out, Nl[0]);
}
